// Network_83554293776835
// MI455X (gfx1250) — matrix operand order repaired and checked by reading, not compiled, not run on hardware
//
#include <hip/hip_runtime.h>
#include <hip/hip_bf16.h>

// ---------------------------------------------------------------------------
// Bidirectional 2-layer GRU + FC for MI455X (gfx1250, wave32, WMMA f16->f32)
// ---------------------------------------------------------------------------
#define Bb   64
#define INC  69
#define Tt   1000
#define Hh   512
#define OUTD 12
#define Gg   1536           // 3*H
#define NTG  96             // Gg/16 n-tiles
#define KPAD0 96            // pad 69 -> 96 (3 k-tiles of 32)

typedef _Float16 v16h __attribute__((ext_vector_type(16)));
typedef _Float16 h8   __attribute__((ext_vector_type(8)));
typedef float    v8f  __attribute__((ext_vector_type(8)));

// B-matrix (32x16 f16): lane L: col N = L%16 ; e<8 -> K = 8*(L/16) + e, e>=8 -> K = 16 + 8*(L/16) + (e-8)
__device__ __forceinline__ int kmapB(int e, int lane) {
    return ((lane >> 4) << 3) + e + ((e >> 3) << 3);
}

// Load A fragment from row-major f16 [.. x ld], tile base = (row0, k0) ptr.
// A layout (ISA 7.12.2): lane L: row M=L%16, s=L/16, e<8 -> K=8s+e, e>=8 -> K=16+8s+(e-8)
// => two contiguous 16B loads per lane.
__device__ __forceinline__ v16h load_frag_A(const _Float16* tile, int ld, int lane) {
    const int m = lane & 15, s = lane >> 4;
    const _Float16* p = tile + (size_t)m * ld + (s << 3);
    h8 lo = *(const h8*)(p);
    h8 hi = *(const h8*)(p + 16);
    v16h r;
#pragma unroll
    for (int i = 0; i < 8; ++i) { r[i] = lo[i]; r[i + 8] = hi[i]; }
    return r;
}

// Load B fragment from pre-swizzled buffer: frag (kt, nt) -> 32 lanes x 16 halves.
__device__ __forceinline__ v16h load_frag_B(const _Float16* Bp, int kt, int nt, int lane) {
    const _Float16* p = Bp + ((((size_t)kt * NTG + nt) << 5) + lane) * 16;
    return *(const v16h*)p;
}

__device__ __forceinline__ v8f wmma_f16(v16h a, v16h b, v8f c) {
    return __builtin_amdgcn_wmma_f32_16x16x32_f16(false, a, false, b, (short)0, c, false, false);
}

// ---------------------------------------------------------------------------
// Pack a weight matrix w[G=1536][Kin] (row-major, gate-major) into per-lane
// WMMA B-fragment layout over Kpad = KT*32 (zero padded).
// ---------------------------------------------------------------------------
__global__ void pack_b_frags(const float* __restrict__ w, _Float16* __restrict__ out,
                             int Kin, int KT) {
    int idx = blockIdx.x * blockDim.x + threadIdx.x;      // one thread = (kt,nt,lane)
    int total = KT * NTG * 32;
    if (idx >= total) return;
    int lane = idx & 31;
    int nt   = (idx >> 5) % NTG;
    int kt   = idx / (NTG * 32);
    int n    = nt * 16 + (lane & 15);
    _Float16* o = out + (size_t)idx * 16;
#pragma unroll
    for (int e = 0; e < 16; ++e) {
        int k = kt * 32 + kmapB(e, lane);
        float v = (k < Kin) ? w[(size_t)n * Kin + k] : 0.0f;
        o[e] = (_Float16)v;
    }
}

// ---------------------------------------------------------------------------
// x (B, IN, T) f32  ->  A0 f16 [T*B][KPAD0] with row m = t*B + b, zero padded K
// ---------------------------------------------------------------------------
__global__ void cvt_x(const float* __restrict__ x, _Float16* __restrict__ A0) {
    int idx = blockIdx.x * blockDim.x + threadIdx.x;
    const int total = Tt * Bb * KPAD0;
    if (idx >= total) return;
    int k = idx % KPAD0;
    int m = idx / KPAD0;
    int b = m & (Bb - 1);
    int t = m / Bb;
    float v = (k < INC) ? x[((size_t)b * INC + k) * Tt + t] : 0.0f;
    A0[idx] = (_Float16)v;
}

// ---------------------------------------------------------------------------
// WMMA GEMM: C[M][1536] f32 = A[M][KT*32] f16 * Bpacked + bias
// block = 256 (8 waves), WG tile 64(M) x 128(N), wave tile 32x32 (2x2 frags).
// grid = (M/64, 12)
// ---------------------------------------------------------------------------
__global__ void gemm_xp(const _Float16* __restrict__ A, int lda, int KT,
                        const _Float16* __restrict__ Bp,
                        const float* __restrict__ bias,
                        float* __restrict__ C) {
    const int lane = threadIdx.x & 31;
    const int wid  = threadIdx.x >> 5;
    const int mt0  = (wid >> 2) << 1;          // 0 or 2
    const int nt0  = (wid & 3) << 1;           // 0,2,4,6 (local n-tiles)
    const int mbase = blockIdx.x << 6;
    const int nbase = blockIdx.y << 7;
    const int ntg0  = (nbase >> 4) + nt0;      // global n-tile index for Bp

    v8f c00 = {}, c01 = {}, c10 = {}, c11 = {};
    for (int kt = 0; kt < KT; ++kt) {
        const _Float16* at0 = A + (size_t)(mbase + ((mt0    ) << 4)) * lda + kt * 32;
        const _Float16* at1 = A + (size_t)(mbase + ((mt0 + 1) << 4)) * lda + kt * 32;
        v16h a0 = load_frag_A(at0, lda, lane);
        v16h a1 = load_frag_A(at1, lda, lane);
        v16h b0 = load_frag_B(Bp, kt, ntg0,     lane);
        v16h b1 = load_frag_B(Bp, kt, ntg0 + 1, lane);
        c00 = wmma_f16(a0, b0, c00);
        c01 = wmma_f16(a0, b1, c01);
        c10 = wmma_f16(a1, b0, c10);
        c11 = wmma_f16(a1, b1, c11);
    }
    // store: lane L -> col = L%16 ; reg r -> row = r + 8*(L/16)
    const int s = lane >> 4;
    const int nc = lane & 15;
#pragma unroll
    for (int half = 0; half < 4; ++half) {
        v8f acc = (half == 0) ? c00 : (half == 1) ? c01 : (half == 2) ? c10 : c11;
        int mt = mt0 + (half >> 1);
        int ntl = nt0 + (half & 1);
        int n = nbase + (ntl << 4) + nc;
        float bv = bias[n];
#pragma unroll
        for (int r = 0; r < 8; ++r) {
            int m = mbase + (mt << 4) + r + (s << 3);
            C[(size_t)m * Gg + n] = acc[r] + bv;
        }
    }
}

// ---------------------------------------------------------------------------
// GRU nonlinear update for one m-tile of accumulators.
// ---------------------------------------------------------------------------
__device__ __forceinline__ void gru_update(
    int mt, v8f cr, v8f cz, v8f cn,
    const float* __restrict__ xp, int tt, int rowBase,
    int jcol, int s, float bhr, float bhz, float bhn,
    const _Float16* __restrict__ h, _Float16* __restrict__ hn,
    _Float16* __restrict__ out, int dir, int storeAll) {
#pragma unroll
    for (int r = 0; r < 8; ++r) {
        const int b = (mt << 4) + r + (s << 3);
        const size_t xrow = ((size_t)(tt - rowBase) * Bb + b) * Gg;
        const float xr = xp[xrow + jcol];
        const float xz = xp[xrow + Hh + jcol];
        const float xn = xp[xrow + 2 * Hh + jcol];
        const float rg = 1.0f / (1.0f + __expf(-(xr + cr[r] + bhr)));
        const float zg = 1.0f / (1.0f + __expf(-(xz + cz[r] + bhz)));
        const float ng = tanhf(xn + rg * (cn[r] + bhn));
        const float hp = (float)h[(size_t)b * Hh + jcol];
        const float hv = (1.0f - zg) * ng + zg * hp;
        hn[(size_t)b * Hh + jcol] = (_Float16)hv;
        if (storeAll) {
            out[((size_t)tt * Bb + b) * (2 * Hh) + dir * Hh + jcol] = (_Float16)hv;
        } else if (tt == Tt - 1) {
            out[(size_t)b * (2 * Hh) + dir * Hh + jcol] = (_Float16)hv;
        }
    }
}

// ---------------------------------------------------------------------------
// Persistent GRU recurrence. grid.x = 2 (dir 0 = fwd, 1 = bwd),
// block = 1024 threads = 32 waves. Wave w owns hidden cols [16w, 16w+16):
// gate n-tiles {w, 32+w, 64+w}. h double-buffered in LDS (2 x 64x512 f16).
// kt-outer loop with 12 live accumulators: each Whh B-fragment is loaded from
// L2 exactly once per step; A fragments come from LDS (ds_load_b128).
// Next step's xp slab (384 KB) is prefetched (global_prefetch_b8) at step start
// to overlap its fetch with the 6144 WMMAs of the current step.
// ---------------------------------------------------------------------------
__global__ void gru_seq(const float* __restrict__ xp0, const float* __restrict__ xp1,
                        int rowBase0, int rowBase1,
                        const _Float16* __restrict__ whh0, const _Float16* __restrict__ whh1,
                        const float* __restrict__ bhh0, const float* __restrict__ bhh1,
                        _Float16* __restrict__ out,
                        int nsteps0, int nsteps1, int storeAll) {
    extern __shared__ _Float16 hbuf[];                 // 2 * 64 * 512
    const int dir = blockIdx.x;
    const float*     xp      = dir ? xp1 : xp0;
    const int        rowBase = dir ? rowBase1 : rowBase0;
    const _Float16*  whh     = dir ? whh1 : whh0;
    const float*     bhh     = dir ? bhh1 : bhh0;
    const int        nsteps  = dir ? nsteps1 : nsteps0;
    const bool       rev     = (dir == 1);

    const int tid  = threadIdx.x;
    const int lane = tid & 31;
    const int w    = tid >> 5;                         // 0..31 == j-tile
    const int s    = lane >> 4;
    const int jcol = (w << 4) + (lane & 15);

    // h0 = 0
    for (int i = tid; i < Bb * Hh; i += 1024) hbuf[i] = (_Float16)0;
    __syncthreads();

    const float bhr = bhh[jcol];
    const float bhz = bhh[Hh + jcol];
    const float bhn = bhh[2 * Hh + jcol];

    int cur = 0;
    for (int step = 0; step < nsteps; ++step) {
        const int tt = rev ? (Tt - 1 - step) : step;
        const _Float16* h  = hbuf + cur * (Bb * Hh);
        _Float16*       hn = hbuf + (1 - cur) * (Bb * Hh);

        // Prefetch next step's xp slab: 64 rows x 6 KB = 6144 64B-lines,
        // 6 lines per thread.
        if (step + 1 < nsteps) {
            const int ttn = rev ? (Tt - 2 - step) : (step + 1);
            const char* slab = (const char*)(xp + (size_t)(ttn - rowBase) * Bb * Gg);
#pragma unroll
            for (int i = 0; i < 6; ++i) {
                __builtin_prefetch(slab + ((size_t)tid * 6 + i) * 64, 0, 3);
            }
        }

        v8f cr0 = {}, cr1 = {}, cr2 = {}, cr3 = {};
        v8f cz0 = {}, cz1 = {}, cz2 = {}, cz3 = {};
        v8f cn0 = {}, cn1 = {}, cn2 = {}, cn3 = {};
#pragma unroll 2
        for (int kt = 0; kt < 16; ++kt) {
            v16h br = load_frag_B(whh, kt, w,      lane);
            v16h bz = load_frag_B(whh, kt, 32 + w, lane);
            v16h bn = load_frag_B(whh, kt, 64 + w, lane);
            v16h a0 = load_frag_A(h + (size_t)( 0) * Hh + kt * 32, Hh, lane);
            v16h a1 = load_frag_A(h + (size_t)(16) * Hh + kt * 32, Hh, lane);
            v16h a2 = load_frag_A(h + (size_t)(32) * Hh + kt * 32, Hh, lane);
            v16h a3 = load_frag_A(h + (size_t)(48) * Hh + kt * 32, Hh, lane);
            cr0 = wmma_f16(a0, br, cr0);  cz0 = wmma_f16(a0, bz, cz0);  cn0 = wmma_f16(a0, bn, cn0);
            cr1 = wmma_f16(a1, br, cr1);  cz1 = wmma_f16(a1, bz, cz1);  cn1 = wmma_f16(a1, bn, cn1);
            cr2 = wmma_f16(a2, br, cr2);  cz2 = wmma_f16(a2, bz, cz2);  cn2 = wmma_f16(a2, bn, cn2);
            cr3 = wmma_f16(a3, br, cr3);  cz3 = wmma_f16(a3, bz, cz3);  cn3 = wmma_f16(a3, bn, cn3);
        }
        gru_update(0, cr0, cz0, cn0, xp, tt, rowBase, jcol, s, bhr, bhz, bhn, h, hn, out, dir, storeAll);
        gru_update(1, cr1, cz1, cn1, xp, tt, rowBase, jcol, s, bhr, bhz, bhn, h, hn, out, dir, storeAll);
        gru_update(2, cr2, cz2, cn2, xp, tt, rowBase, jcol, s, bhr, bhz, bhn, h, hn, out, dir, storeAll);
        gru_update(3, cr3, cz3, cn3, xp, tt, rowBase, jcol, s, bhr, bhz, bhn, h, hn, out, dir, storeAll);
        __syncthreads();
        cur ^= 1;
    }
}

// ---------------------------------------------------------------------------
// FC: out[b][o] = l1last[b][:1024] . fc_w[o][:1024] + fc_b[o]   (tiny)
// ---------------------------------------------------------------------------
__global__ void fc_kernel(const _Float16* __restrict__ l1last,
                          const float* __restrict__ fcw,
                          const float* __restrict__ fcb,
                          float* __restrict__ out) {
    int idx = blockIdx.x * blockDim.x + threadIdx.x;
    if (idx >= Bb * OUTD) return;
    int b = idx / OUTD, o = idx % OUTD;
    float acc = fcb[o];
    for (int k = 0; k < 2 * Hh; ++k)
        acc += (float)l1last[(size_t)b * (2 * Hh) + k] * fcw[(size_t)o * (2 * Hh) + k];
    out[(size_t)b * OUTD + o] = acc;
}

// ---------------------------------------------------------------------------
extern "C" void kernel_launch(void* const* d_in, const int* in_sizes, int n_in,
                              void* d_out, int out_size, void* d_ws, size_t ws_size,
                              hipStream_t stream) {
    (void)in_sizes; (void)n_in; (void)out_size; (void)ws_size;
    const float* x        = (const float*)d_in[0];
    const float* w_ih_l0f = (const float*)d_in[1];
    const float* w_hh_l0f = (const float*)d_in[2];
    const float* b_ih_l0f = (const float*)d_in[3];
    const float* b_hh_l0f = (const float*)d_in[4];
    const float* w_ih_l0b = (const float*)d_in[5];
    const float* w_hh_l0b = (const float*)d_in[6];
    const float* b_ih_l0b = (const float*)d_in[7];
    const float* b_hh_l0b = (const float*)d_in[8];
    const float* w_ih_l1f = (const float*)d_in[9];
    const float* w_hh_l1f = (const float*)d_in[10];
    const float* b_ih_l1f = (const float*)d_in[11];
    const float* b_hh_l1f = (const float*)d_in[12];
    const float* w_ih_l1b = (const float*)d_in[13];
    const float* w_hh_l1b = (const float*)d_in[14];
    const float* b_ih_l1b = (const float*)d_in[15];
    const float* b_hh_l1b = (const float*)d_in[16];
    const float* fc_w     = (const float*)d_in[17];
    const float* fc_b     = (const float*)d_in[18];

    // ---- workspace carve (256B aligned) ----
    char* ws = (char*)d_ws;
    size_t off = 0;
    auto carve = [&](size_t bytes) -> char* {
        char* p = ws + off;
        off = (off + bytes + 255) & ~(size_t)255;
        return p;
    };
    const size_t fragBytes = (size_t)NTG * 32 * 16 * 2;           // per k-tile
    _Float16* A0      = (_Float16*)carve((size_t)Tt * Bb * KPAD0 * 2);
    _Float16* l0out   = (_Float16*)carve((size_t)Tt * Bb * 2 * Hh * 2);
    float*    xpF     = (float*)carve((size_t)Tt * Bb * Gg * 4);
    float*    xpB     = (float*)carve((size_t)Tt * Bb * Gg * 4);
    float*    xp1b    = (float*)carve((size_t)Bb * Gg * 4);
    _Float16* wih0f_p = (_Float16*)carve(3  * fragBytes);
    _Float16* wih0b_p = (_Float16*)carve(3  * fragBytes);
    _Float16* whh0f_p = (_Float16*)carve(16 * fragBytes);
    _Float16* whh0b_p = (_Float16*)carve(16 * fragBytes);
    _Float16* whh1f_p = (_Float16*)carve(16 * fragBytes);
    _Float16* whh1b_p = (_Float16*)carve(16 * fragBytes);
    _Float16* wih1f_p = (_Float16*)carve(32 * fragBytes);
    _Float16* wih1b_p = (_Float16*)carve(32 * fragBytes);
    _Float16* l1last  = (_Float16*)carve((size_t)Bb * 2 * Hh * 2);

    // ---- 1. pack weights into WMMA B-fragment layout ----
    auto packLaunch = [&](const float* w, _Float16* dst, int Kin, int KT) {
        int total = KT * NTG * 32;
        pack_b_frags<<<dim3((total + 255) / 256), dim3(256), 0, stream>>>(w, dst, Kin, KT);
    };
    packLaunch(w_ih_l0f, wih0f_p, INC,    3);
    packLaunch(w_ih_l0b, wih0b_p, INC,    3);
    packLaunch(w_hh_l0f, whh0f_p, Hh,     16);
    packLaunch(w_hh_l0b, whh0b_p, Hh,     16);
    packLaunch(w_hh_l1f, whh1f_p, Hh,     16);
    packLaunch(w_hh_l1b, whh1b_p, Hh,     16);
    packLaunch(w_ih_l1f, wih1f_p, 2 * Hh, 32);
    packLaunch(w_ih_l1b, wih1b_p, 2 * Hh, 32);

    // ---- 2. x -> f16 A matrix (row = t*B + b) ----
    {
        int total = Tt * Bb * KPAD0;
        cvt_x<<<dim3((total + 255) / 256), dim3(256), 0, stream>>>(x, A0);
    }

    // ---- 3. layer-0 input projections (+ b_ih) ----
    gemm_xp<<<dim3(Tt * Bb / 64, Gg / 128), dim3(256), 0, stream>>>(
        A0, KPAD0, 3, wih0f_p, b_ih_l0f, xpF);
    gemm_xp<<<dim3(Tt * Bb / 64, Gg / 128), dim3(256), 0, stream>>>(
        A0, KPAD0, 3, wih0b_p, b_ih_l0b, xpB);

    // ---- 4. layer-0 recurrence (both directions concurrently) ----
    const size_t ldsBytes = 2 * (size_t)Bb * Hh * sizeof(_Float16);  // 128 KB
    gru_seq<<<dim3(2), dim3(1024), ldsBytes, stream>>>(
        xpF, xpB, 0, 0, whh0f_p, whh0b_p, b_hh_l0f, b_hh_l0b,
        l0out, Tt, Tt, /*storeAll=*/1);

    // ---- 5. layer-1 input projections ----
    // forward: all T rows; backward: only t = T-1 (its output at t=T-1 is the
    // hidden state after a single scan step — the only one the FC consumes).
    gemm_xp<<<dim3(Tt * Bb / 64, Gg / 128), dim3(256), 0, stream>>>(
        l0out, 2 * Hh, 32, wih1f_p, b_ih_l1f, xpF);
    gemm_xp<<<dim3(1, Gg / 128), dim3(256), 0, stream>>>(
        l0out + (size_t)(Tt - 1) * Bb * 2 * Hh, 2 * Hh, 32, wih1b_p, b_ih_l1b, xp1b);

    // ---- 6. layer-1 recurrence: fwd 1000 steps, bwd 1 step ----
    gru_seq<<<dim3(2), dim3(1024), ldsBytes, stream>>>(
        xpF, xp1b, 0, Tt - 1, whh1f_p, whh1b_p, b_hh_l1f, b_hh_l1b,
        l1last, Tt, 1, /*storeAll=*/0);

    // ---- 7. FC head ----
    fc_kernel<<<dim3((Bb * OUTD + 255) / 256), dim3(256), 0, stream>>>(
        l1last, fc_w, fc_b, (float*)d_out);
}
